// Encoder_9216999817313
// MI455X (gfx1250) — matrix operand order repaired and checked by reading, not compiled, not run on hardware
//
#include <hip/hip_runtime.h>

// ---------------------------------------------------------------------------
// EGNN (4 layers) for MI455X / gfx1250: wave32 + v_wmma_f32_16x16x32_bf16.
//  - h_bf16 (6.4MB) is L2-resident -> random edge gathers are L2 traffic.
//  - HBM traffic ~ streaming extra[E][32] + agg atomics + outputs, far under
//    the 23.3 TB/s roofline; so minimize instructions/byte:
//      * weights pre-swizzled into B-fragment order, read straight from
//        global (WGP$-hot), no LDS staging, no __syncthreads
//      * e_in/n_in staged to LDS with 16B vector copies only
//      * A fragments = 2x ds_load_b128; B fragments = 1x 32B global load
//      * SiLU via v_rcp_f32 instead of IEEE division expansion
// ---------------------------------------------------------------------------

#define DEV __device__ __forceinline__

typedef __attribute__((ext_vector_type(16))) __bf16         v16bf;
typedef __attribute__((ext_vector_type(16))) unsigned short v16us;
typedef __attribute__((ext_vector_type(8)))  unsigned short v8us;
typedef __attribute__((ext_vector_type(8)))  float          v8f;
typedef __attribute__((ext_vector_type(4)))  unsigned int   v4u;
typedef __attribute__((ext_vector_type(8)))  unsigned int   v8u;

constexpr int N_NODES = 50000;
constexpr int N_EDGES = 800000;
constexpr int C_IN    = 32;
constexpr int C_H     = 64;
constexpr int C_ED    = 16;
constexpr int C_PD    = 3;

DEV unsigned short f2bf(float f) {
  unsigned int u = __float_as_uint(f);
  unsigned int r = u + 0x7FFFu + ((u >> 16) & 1u);   // RNE
  return (unsigned short)(r >> 16);
}

DEV float silu(float x) {
  return x * __builtin_amdgcn_rcpf(1.0f + __expf(-x));
}

// CDNA5 split dep counter: wait for this wave's outstanding LDS ops.
DEV void wait_ds() { asm volatile("s_wait_dscnt 0" ::: "memory"); }

// A fragment (16x32 bf16, row-major tile in LDS): lane m=lane&15,
// K-halves split by lane half -> two contiguous 16B LDS loads.
DEV v16bf load_a(const unsigned short* __restrict__ lds, int pitch,
                 int m, int kbase, int hi) {
  const unsigned short* p = lds + m * pitch + kbase + hi * 8;
  v4u lo = *(const v4u*)p;
  v4u h4 = *(const v4u*)(p + 16);
  v8u w;
  w.lo = lo;
  w.hi = h4;
  return __builtin_bit_cast(v16bf, w);
}

// B fragment from pre-swizzled weights: [frag][lane][16 bf16] contiguous.
DEV v16bf load_bfrag(const unsigned short* __restrict__ W, int frag, int lane) {
  return __builtin_bit_cast(
      v16bf, *(const v8u*)(W + ((size_t)frag * 32 + lane) * 16));
}

// ---------------------------------------------------------------------------
// Weight convert f32 -> bf16 in B-fragment-swizzled order.
// frag = ks*4+nt; element j of lane: k = ks*32 + (j>=8)*16 + (lane>=16)*8 + (j&7),
//                                    n = nt*16 + (lane&15). Zero pad k>=rows.
// ---------------------------------------------------------------------------
__global__ void convert_swz(const float* __restrict__ W,
                            unsigned short* __restrict__ dst,
                            int srcRows, int totalFrags) {
  int o = blockIdx.x * blockDim.x + threadIdx.x;
  if (o >= totalFrags * 512) return;
  int frag   = o >> 9;
  int within = o & 511;
  int lane = within >> 4;
  int j    = within & 15;
  int ks = frag >> 2;
  int nt = frag & 3;
  int k = ks * 32 + ((j >> 3) << 4) + ((lane >> 4) << 3) + (j & 7);
  int n = nt * 16 + (lane & 15);
  dst[o] = (k < srcRows) ? f2bf(W[k * 64 + n]) : (unsigned short)0;
}

// Plain elementwise f32 -> bf16 (for h).
__global__ void convert_bf(const float* __restrict__ src,
                           unsigned short* __restrict__ dst, int n) {
  int i = blockIdx.x * blockDim.x + threadIdx.x;
  if (i < n) dst[i] = f2bf(src[i]);
}

// ---------------------------------------------------------------------------
// Per-edge invariant block: extra[e][32] = [d2 | edge_attr(16) | 0-pad(15)].
// ---------------------------------------------------------------------------
__global__ void build_extra(const float* __restrict__ pos,
                            const int* __restrict__ ei,
                            const float* __restrict__ ea,
                            unsigned short* __restrict__ extra) {
  int e = blockIdx.x * blockDim.x + threadIdx.x;
  if (e >= N_EDGES) return;
  int r = ei[e];
  int c = ei[N_EDGES + e];
  float dx = pos[r * 3 + 0] - pos[c * 3 + 0];
  float dy = pos[r * 3 + 1] - pos[c * 3 + 1];
  float dz = pos[r * 3 + 2] - pos[c * 3 + 2];
  unsigned short* o = extra + (size_t)e * 32;
  o[0] = f2bf(dx * dx + dy * dy + dz * dz);
#pragma unroll
  for (int j = 0; j < C_ED; ++j) o[1 + j] = f2bf(ea[(size_t)e * C_ED + j]);
#pragma unroll
  for (int j = 17; j < 32; ++j) o[j] = 0;
}

// ---------------------------------------------------------------------------
// Edge kernel: one wave per 16-edge tile; 8 independent waves per block.
// ---------------------------------------------------------------------------
template <int DIN, int KPADE>
__global__ __launch_bounds__(256) void edge_kernel(
    const unsigned short* __restrict__ hbf,
    const unsigned short* __restrict__ extra,
    const int* __restrict__ edge_index,
    const unsigned short* __restrict__ We1s, const float* __restrict__ be1,
    const unsigned short* __restrict__ We2s, const float* __restrict__ be2,
    float* __restrict__ agg) {
  __shared__ alignas(16) unsigned short sEin[8][16 * KPADE];
  __shared__ alignas(16) unsigned short sM1[8][16 * 64];
  __shared__ int sRow[8][16];
  __shared__ int sCol[8][16];

  int wave = threadIdx.x >> 5;
  int lane = threadIdx.x & 31;
  int tile = blockIdx.x * 8 + wave;
  if (tile * 16 >= N_EDGES) return;
  int e0 = tile * 16;

  unsigned short* ein = sEin[wave];
  unsigned short* m1  = sM1[wave];
  int* rw = sRow[wave];
  int* cl = sCol[wave];

  if (lane < 16) rw[lane] = edge_index[e0 + lane];
  else           cl[lane - 16] = edge_index[N_EDGES + e0 + lane - 16];
  wait_ds();

  // Stage e_in = [h[row] | h[col] | extra] as 16B chunks only.
  constexpr int CPE  = DIN / 4 + 4;        // chunks (8 bf16) per edge row
  constexpr int ITER = 16 * CPE / 32;
#pragma unroll
  for (int t = 0; t < ITER; ++t) {
    int flat = lane + 32 * t;
    int e = flat / CPE;
    int c = flat - e * CPE;
    const unsigned short* src;
    if (c < DIN / 8)      src = hbf + (size_t)rw[e] * DIN + c * 8;
    else if (c < DIN / 4) src = hbf + (size_t)cl[e] * DIN + (c - DIN / 8) * 8;
    else                  src = extra + (size_t)(e0 + e) * 32 + (c - DIN / 4) * 8;
    *(v4u*)&ein[e * KPADE + c * 8] = *(const v4u*)src;
  }
  wait_ds();

  int nl = lane & 15;
  int hi = lane >> 4;

  // GEMM1: [16 x KPADE] @ We1 + be1, SiLU -> m1 (bf16 LDS)
  for (int nt = 0; nt < 4; ++nt) {
    v8f c;
    float bias = be1[nt * 16 + nl];
#pragma unroll
    for (int r = 0; r < 8; ++r) c[r] = bias;
#pragma unroll
    for (int ks = 0; ks < KPADE / 32; ++ks) {
      v16bf a = load_a(ein, KPADE, nl, ks * 32, hi);
      v16bf b = load_bfrag(We1s, ks * 4 + nt, lane);
      c = __builtin_amdgcn_wmma_f32_16x16x32_bf16(false, a, false, b,
                                                  (short)0, c, false, false);
    }
#pragma unroll
    for (int r = 0; r < 8; ++r)
      m1[(hi * 8 + r) * 64 + nt * 16 + nl] = f2bf(silu(c[r]));
  }
  wait_ds();

  // GEMM2: [16 x 64] @ We2 + be2, SiLU, atomic scatter-add into agg[row].
  for (int nt = 0; nt < 4; ++nt) {
    v8f c;
    float bias = be2[nt * 16 + nl];
#pragma unroll
    for (int r = 0; r < 8; ++r) c[r] = bias;
#pragma unroll
    for (int ks = 0; ks < 2; ++ks) {
      v16bf a = load_a(m1, 64, nl, ks * 32, hi);
      v16bf b = load_bfrag(We2s, ks * 4 + nt, lane);
      c = __builtin_amdgcn_wmma_f32_16x16x32_bf16(false, a, false, b,
                                                  (short)0, c, false, false);
    }
#pragma unroll
    for (int r = 0; r < 8; ++r)
      atomicAdd(&agg[(size_t)rw[hi * 8 + r] * 64 + nt * 16 + nl], silu(c[r]));
  }
}

// ---------------------------------------------------------------------------
// Node kernel: one wave per 16-node tile; n_in = [h | agg].
// ---------------------------------------------------------------------------
template <int DIN, int KPADN>
__global__ __launch_bounds__(256) void node_kernel(
    const unsigned short* __restrict__ hbf,
    const float* __restrict__ agg,
    const unsigned short* __restrict__ Wh1s, const float* __restrict__ bh1,
    const unsigned short* __restrict__ Wh2s, const float* __restrict__ bh2,
    float* __restrict__ out) {
  __shared__ alignas(16) unsigned short sNin[8][16 * KPADN];
  __shared__ alignas(16) unsigned short sM1[8][16 * 64];

  int wave = threadIdx.x >> 5;
  int lane = threadIdx.x & 31;
  int tile = blockIdx.x * 8 + wave;
  if (tile * 16 >= N_NODES) return;
  int n0 = tile * 16;

  unsigned short* nin = sNin[wave];
  unsigned short* m1  = sM1[wave];

  // Stage n_in: copy h (bf16 16B chunks) + convert agg f32->bf16 in 8-chunks.
  constexpr int CPN  = DIN / 8 + 8;
  constexpr int ITER = 16 * CPN / 32;
#pragma unroll
  for (int t = 0; t < ITER; ++t) {
    int flat = lane + 32 * t;
    int e = flat / CPN;
    int c = flat - e * CPN;
    if (c < DIN / 8) {
      *(v4u*)&nin[e * KPADN + c * 8] =
          *(const v4u*)(hbf + (size_t)(n0 + e) * DIN + c * 8);
    } else {
      int cc = c - DIN / 8;
      const float* s = agg + (size_t)(n0 + e) * 64 + cc * 8;
      v8us o;
#pragma unroll
      for (int j = 0; j < 8; ++j) o[j] = f2bf(s[j]);
      *(v8us*)&nin[e * KPADN + DIN + cc * 8] = o;
    }
  }
  wait_ds();

  int nl = lane & 15;
  int hi = lane >> 4;

  for (int nt = 0; nt < 4; ++nt) {
    v8f c;
    float bias = bh1[nt * 16 + nl];
#pragma unroll
    for (int r = 0; r < 8; ++r) c[r] = bias;
#pragma unroll
    for (int ks = 0; ks < KPADN / 32; ++ks) {
      v16bf a = load_a(nin, KPADN, nl, ks * 32, hi);
      v16bf b = load_bfrag(Wh1s, ks * 4 + nt, lane);
      c = __builtin_amdgcn_wmma_f32_16x16x32_bf16(false, a, false, b,
                                                  (short)0, c, false, false);
    }
#pragma unroll
    for (int r = 0; r < 8; ++r)
      m1[(hi * 8 + r) * 64 + nt * 16 + nl] = f2bf(silu(c[r]));
  }
  wait_ds();

  for (int nt = 0; nt < 4; ++nt) {
    v8f c;
    float bias = bh2[nt * 16 + nl];
#pragma unroll
    for (int r = 0; r < 8; ++r) c[r] = bias;
#pragma unroll
    for (int ks = 0; ks < 2; ++ks) {
      v16bf a = load_a(m1, 64, nl, ks * 32, hi);
      v16bf b = load_bfrag(Wh2s, ks * 4 + nt, lane);
      c = __builtin_amdgcn_wmma_f32_16x16x32_bf16(false, a, false, b,
                                                  (short)0, c, false, false);
    }
#pragma unroll
    for (int r = 0; r < 8; ++r)
      out[(size_t)(n0 + hi * 8 + r) * 64 + nt * 16 + nl] = c[r];  // no act
  }
}

// ---------------------------------------------------------------------------
// Host launcher.  d_in: x, pos, edge_index, edge_attr, then per layer
// (We1,be1,We2,be2,Wh1,bh1,Wh2,bh2) x 4.  d_out = [h (N*64) | pos (N*3)].
// ---------------------------------------------------------------------------
extern "C" void kernel_launch(void* const* d_in, const int* in_sizes, int n_in,
                              void* d_out, int out_size, void* d_ws,
                              size_t ws_size, hipStream_t stream) {
  const float* x         = (const float*)d_in[0];
  const float* pos       = (const float*)d_in[1];
  const int*   edgeIdx   = (const int*)d_in[2];
  const float* edge_attr = (const float*)d_in[3];
  auto P = [&](int l, int j) { return (const float*)d_in[4 + l * 8 + j]; };

  // ---- workspace layout (all blocks 32B-multiple sized) ----
  char* w = (char*)d_ws;
  unsigned short* extra = (unsigned short*)w; w += (size_t)N_EDGES * 32 * 2;
  unsigned short* hbf   = (unsigned short*)w; w += (size_t)N_NODES * 64 * 2;
  float* agg = (float*)w; w += (size_t)N_NODES * 64 * 4;
  float* hA  = (float*)w; w += (size_t)N_NODES * 64 * 4;
  float* hB  = (float*)w; w += (size_t)N_NODES * 64 * 4;

  const int KpadE[4] = {96, 160, 160, 160};
  const int KpadN[4] = {96, 128, 128, 128};
  const int rowsE[4] = {81, 145, 145, 145};      // 2*din+1+ED
  const int rowsN[4] = {96, 128, 128, 128};      // din+H (exact)

  unsigned short *We1s[4], *We2s[4], *Wh1s[4], *Wh2s[4];
  for (int l = 0; l < 4; ++l) {
    We1s[l] = (unsigned short*)w; w += (size_t)KpadE[l] * 64 * 2;
    We2s[l] = (unsigned short*)w; w += (size_t)64 * 64 * 2;
    Wh1s[l] = (unsigned short*)w; w += (size_t)KpadN[l] * 64 * 2;
    Wh2s[l] = (unsigned short*)w; w += (size_t)64 * 64 * 2;
  }

  auto swz = [&](const float* src, unsigned short* dst, int rows, int kpad) {
    int frags = (kpad / 32) * 4;
    int total = frags * 512;
    convert_swz<<<(total + 255) / 256, 256, 0, stream>>>(src, dst, rows, frags);
  };

  for (int l = 0; l < 4; ++l) {
    swz(P(l, 0), We1s[l], rowsE[l], KpadE[l]);
    swz(P(l, 2), We2s[l], 64, 64);
    swz(P(l, 4), Wh1s[l], rowsN[l], KpadN[l]);
    swz(P(l, 6), Wh2s[l], 64, 64);
  }
  build_extra<<<(N_EDGES + 255) / 256, 256, 0, stream>>>(pos, edgeIdx,
                                                         edge_attr, extra);

  const int edgeBlocks = N_EDGES / 16 / 8;                 // 6250
  const int nodeBlocks = (N_NODES / 16 + 7) / 8;           // 391

  float* hOut = (float*)d_out;
  const float* hs[4] = {x, hA, hB, hA};

  for (int l = 0; l < 4; ++l) {
    int din = (l == 0) ? C_IN : C_H;
    int nh = N_NODES * din;
    convert_bf<<<(nh + 255) / 256, 256, 0, stream>>>(hs[l], hbf, nh);
    hipMemsetAsync(agg, 0, (size_t)N_NODES * 64 * 4, stream);
    if (l == 0)
      edge_kernel<32, 96><<<edgeBlocks, 256, 0, stream>>>(
          hbf, extra, edgeIdx, We1s[l], P(l, 1), We2s[l], P(l, 3), agg);
    else
      edge_kernel<64, 160><<<edgeBlocks, 256, 0, stream>>>(
          hbf, extra, edgeIdx, We1s[l], P(l, 1), We2s[l], P(l, 3), agg);
    float* dst = (l == 3) ? hOut : ((l % 2 == 0) ? hA : hB);
    if (l == 0)
      node_kernel<32, 96><<<nodeBlocks, 256, 0, stream>>>(
          hbf, agg, Wh1s[l], P(l, 5), Wh2s[l], P(l, 7), dst);
    else
      node_kernel<64, 128><<<nodeBlocks, 256, 0, stream>>>(
          hbf, agg, Wh1s[l], P(l, 5), Wh2s[l], P(l, 7), dst);
  }

  hipMemcpyAsync((float*)d_out + (size_t)N_NODES * 64, pos,
                 (size_t)N_NODES * C_PD * sizeof(float),
                 hipMemcpyDeviceToDevice, stream);
}
